// AdaptiveReLULayer_11982958756442
// MI455X (gfx1250) — matrix operand order repaired and checked by reading, not compiled, not run on hardware
//
#include <hip/hip_runtime.h>
#include <hip/hip_bf16.h>

// ---------------------------------------------------------------------------
// AdaptiveReLULayer: out[b] = leaky_relu( x[b] @ weight[indices[b]] + bias )
// B=2048, N=256, K(IN)=256, M(OUT)=256, C=1024, f32 in/out.
//
// Split-f16 (hi/lo) emulated-f32 GEMM on v_wmma_f32_16x16x32_f16:
//   out ~= x_hi*w_hi + x_hi*w_lo + x_lo*w_hi   (lo*lo term ~2^-22, dropped)
// 3 f16 WMMAs = 3/8 the per-FLOP cost of V_WMMA_F32_16X16X4_F32, keeping the
// kernel near the ~70us HBM roofline (1.3-1.6 GB @ 23.3 TB/s) instead of 3x
// compute-bound.  Double-buffered LDS pipeline: stage panel k+1 (float4
// global loads, f32->hi/lo f16 convert, transposed W) while WMMAs consume
// panel k; one barrier per k-step.
// ---------------------------------------------------------------------------

typedef __attribute__((ext_vector_type(16))) _Float16 v16h;
typedef __attribute__((ext_vector_type(8)))  _Float16 v8h;
typedef __attribute__((ext_vector_type(4)))  _Float16 v4h;
typedef __attribute__((ext_vector_type(8)))  float    v8f;

union Frag16 {
    v16h v;
    v8h  h[2];
};

#define LDSTRIDE 40   // halves per LDS row: 32 data + 8 pad (80B, 16B aligned)

__global__ __launch_bounds__(256)
void adaptive_relu_gemm(const float* __restrict__ x,
                        const int*   __restrict__ indices,
                        const float* __restrict__ weight,
                        const float* __restrict__ bias,
                        float* __restrict__ out)
{
    // Double-buffered staging: x panel 64x32 (hi/lo), w panel transposed
    // [o][k] 256x32 (hi/lo).  2 * 25600 halves = 100 KB (320 KB/WGP on CDNA5).
    __shared__ _Float16 xs_hi[2][64  * LDSTRIDE];
    __shared__ _Float16 xs_lo[2][64  * LDSTRIDE];
    __shared__ _Float16 ws_hi[2][256 * LDSTRIDE];
    __shared__ _Float16 ws_lo[2][256 * LDSTRIDE];

    const int tid  = threadIdx.x;
    const int b    = blockIdx.x >> 2;          // batch
    const int n0   = (blockIdx.x & 3) << 6;    // row slab base (0,64,128,192)

    const int widx = indices[b];
    const float* xb = x      + ((size_t)b * 256 + n0) * 256;
    const float* wb = weight + (size_t)widx * 256 * 256;
    float*       ob = out    + ((size_t)b * 256 + n0) * 256;

    const int lane    = tid & 31;
    const int wave    = tid >> 5;      // 8 waves
    const int halfsel = lane >> 4;     // 0: lanes 0-15, 1: lanes 16-31
    const int l16     = lane & 15;
    const int o_base  = wave * 32;     // each wave owns a 32-wide output stripe

    // ---- panel staging (float4 global loads, hi/lo f16 split) --------------
    auto stage_x = [&](int p, int k0) {
        // 64x32 floats = 512 float4, 2 per thread; 8 float4 per row
#pragma unroll
        for (int i = 0; i < 2; ++i) {
            int e4 = tid + i * 256;            // 0..511
            int n  = e4 >> 3;
            int k  = (e4 & 7) << 2;
            float4 v = *(const float4*)(xb + n * 256 + k0 + k);
            v4h hi, lo;
            hi[0] = (_Float16)v.x;  lo[0] = (_Float16)(v.x - (float)hi[0]);
            hi[1] = (_Float16)v.y;  lo[1] = (_Float16)(v.y - (float)hi[1]);
            hi[2] = (_Float16)v.z;  lo[2] = (_Float16)(v.z - (float)hi[2]);
            hi[3] = (_Float16)v.w;  lo[3] = (_Float16)(v.w - (float)hi[3]);
            *(v4h*)&xs_hi[p][n * LDSTRIDE + k] = hi;   // 8B-aligned b64 store
            *(v4h*)&xs_lo[p][n * LDSTRIDE + k] = lo;
        }
    };
    auto stage_w = [&](int p, int k0) {
        // 32x256 floats = 2048 float4, 8 per thread; 64 float4 per row.
        // Coalesced b128 row reads, transposed [o][k] b16 scatter into LDS.
#pragma unroll
        for (int i = 0; i < 8; ++i) {
            int e4 = tid + i * 256;            // 0..2047
            int r  = e4 >> 6;
            int c  = (e4 & 63) << 2;
            float4 v = *(const float4*)(wb + (size_t)(k0 + r) * 256 + c);
            _Float16 h0 = (_Float16)v.x, h1 = (_Float16)v.y;
            _Float16 h2 = (_Float16)v.z, h3 = (_Float16)v.w;
            ws_hi[p][(c + 0) * LDSTRIDE + r] = h0;
            ws_hi[p][(c + 1) * LDSTRIDE + r] = h1;
            ws_hi[p][(c + 2) * LDSTRIDE + r] = h2;
            ws_hi[p][(c + 3) * LDSTRIDE + r] = h3;
            ws_lo[p][(c + 0) * LDSTRIDE + r] = (_Float16)(v.x - (float)h0);
            ws_lo[p][(c + 1) * LDSTRIDE + r] = (_Float16)(v.y - (float)h1);
            ws_lo[p][(c + 2) * LDSTRIDE + r] = (_Float16)(v.z - (float)h2);
            ws_lo[p][(c + 3) * LDSTRIDE + r] = (_Float16)(v.w - (float)h3);
        }
    };

    v8f acc[4][2];                     // [n-tile][o-tile] 16x16 f32 accumulators
#pragma unroll
    for (int nt = 0; nt < 4; ++nt)
#pragma unroll
        for (int ot = 0; ot < 2; ++ot)
            acc[nt][ot] = (v8f){0.f,0.f,0.f,0.f,0.f,0.f,0.f,0.f};

    // prologue: stage first panel into buffer 0
    stage_x(0, 0);
    stage_w(0, 0);

    int p = 0;
    for (int step = 0; step < 8; ++step) {
        __syncthreads();   // buf[p] staged; buf[p^1] readers from step-1 done

        // stage next panel into the other buffer while we compute
        if (step < 7) {
            int k0n = (step + 1) * 32;
            stage_x(p ^ 1, k0n);
            stage_w(p ^ 1, k0n);
        }

        // ---- build A fragments (16x32 f16 per ISA layout) -------------------
        // lane L<16: row L, halves 0-7 = K 0-7, halves 8-15 = K 16-23
        // lane L+16: row L, halves 0-7 = K 8-15, halves 8-15 = K 24-31
        Frag16 ahi[4], alo[4];
        const int kq0 = halfsel * 8;
#pragma unroll
        for (int nt = 0; nt < 4; ++nt) {
            int row = nt * 16 + l16;
            const _Float16* ph = &xs_hi[p][row * LDSTRIDE];
            const _Float16* pl = &xs_lo[p][row * LDSTRIDE];
            ahi[nt].h[0] = *(const v8h*)(ph + kq0);
            ahi[nt].h[1] = *(const v8h*)(ph + 16 + kq0);
            alo[nt].h[0] = *(const v8h*)(pl + kq0);
            alo[nt].h[1] = *(const v8h*)(pl + 16 + kq0);
        }
        // ---- build B fragments (32x16 f16 per ISA layout) -------------------
        // lane = column (l16); lanes 0-15 hold K 0-7,16-23, lanes 16-31 hold K 8-15,24-31
        Frag16 bhi[2], blo[2];
#pragma unroll
        for (int ot = 0; ot < 2; ++ot) {
            int col = o_base + ot * 16 + l16;
            const _Float16* ph = &ws_hi[p][col * LDSTRIDE + halfsel * 8];
            const _Float16* pl = &ws_lo[p][col * LDSTRIDE + halfsel * 8];
            bhi[ot].h[0] = *(const v8h*)(ph);
            bhi[ot].h[1] = *(const v8h*)(ph + 16);
            blo[ot].h[0] = *(const v8h*)(pl);
            blo[ot].h[1] = *(const v8h*)(pl + 16);
        }

        // ---- 3-product split-f16 accumulation (drop lo*lo term) -------------
#pragma unroll
        for (int nt = 0; nt < 4; ++nt)
#pragma unroll
            for (int ot = 0; ot < 2; ++ot) {
                acc[nt][ot] = __builtin_amdgcn_wmma_f32_16x16x32_f16(
                    false, ahi[nt].v, false, bhi[ot].v, (short)0, acc[nt][ot], false, false);
                acc[nt][ot] = __builtin_amdgcn_wmma_f32_16x16x32_f16(
                    false, ahi[nt].v, false, blo[ot].v, (short)0, acc[nt][ot], false, false);
                acc[nt][ot] = __builtin_amdgcn_wmma_f32_16x16x32_f16(
                    false, alo[nt].v, false, bhi[ot].v, (short)0, acc[nt][ot], false, false);
            }

        p ^= 1;
    }

    // ---- epilogue: bias + LeakyReLU(0.2), store per C/D VGPR layout ---------
    // acc element i of lane: row = ntile*16 + i + 8*halfsel, col = otile*16+l16
#pragma unroll
    for (int ot = 0; ot < 2; ++ot) {
        int col  = o_base + ot * 16 + l16;
        float bv = bias[col];
#pragma unroll
        for (int nt = 0; nt < 4; ++nt) {
            int rbase = nt * 16 + halfsel * 8;
#pragma unroll
            for (int i = 0; i < 8; ++i) {
                float v = acc[nt][ot][i] + bv;
                v = (v >= 0.0f) ? v : 0.2f * v;
                ob[(size_t)(rbase + i) * 256 + col] = v;
            }
        }
    }
}

extern "C" void kernel_launch(void* const* d_in, const int* in_sizes, int n_in,
                              void* d_out, int out_size, void* d_ws, size_t ws_size,
                              hipStream_t stream)
{
    const float* x       = (const float*)d_in[0];   // [2048,256,256]
    const int*   indices = (const int*)  d_in[1];   // [2048]
    const float* weight  = (const float*)d_in[2];   // [1024,256,256]
    const float* bias    = (const float*)d_in[3];   // [1,1,256]
    float*       out     = (float*)d_out;           // [2048,256,256]

    (void)in_sizes; (void)n_in; (void)out_size; (void)d_ws; (void)ws_size;

    const int B = 2048;
    dim3 grid(B * 4);   // 4 row-slabs of 64 per batch
    dim3 block(256);    // 8 wave32
    adaptive_relu_gemm<<<grid, block, 0, stream>>>(x, indices, weight, bias, out);
}
